// EncoderLayer_2551210574137
// MI455X (gfx1250) — matrix operand order repaired and checked by reading, not compiled, not run on hardware
//
#include <hip/hip_runtime.h>

#define DEV __device__ __forceinline__

typedef __attribute__((ext_vector_type(16))) __bf16 v16bf;
typedef __attribute__((ext_vector_type(8)))  float  v8f;
typedef __attribute__((ext_vector_type(4)))  unsigned int uint32x4;
typedef __attribute__((ext_vector_type(8)))  int int32x8;
typedef __attribute__((ext_vector_type(4)))  int int32x4;

constexpr int   Bc = 2, Sc = 2048, Dc = 1024, Hc = 16, DKc = 64, DFFc = 4096;
constexpr int   BS = Bc * Sc;          // 4096 rows
constexpr float EPSc = 1e-6f;
constexpr float NEGc = -1e9f;

union AFrag { v16bf v; unsigned int u[8]; unsigned short s[16]; };
union CFrag { v8f   v; float f[8]; };

DEV unsigned short f2bf(float f) {
  unsigned int u = __builtin_bit_cast(unsigned int, f);
  u = u + 0x7fffu + ((u >> 16) & 1u);          // round-to-nearest-even
  return (unsigned short)(u >> 16);
}

// A-matrix 16x32 bf16 fragment K-index map (wave32, contiguous: k = 16*half + 2*j)
DEV int a_kmap(int j, int half) {
  return half * 16 + 2 * j;
}

#if __has_builtin(__builtin_amdgcn_tensor_load_to_lds)
#define USE_TDM 1
// 2D tile load via Tensor Data Mover (D# per ISA cdna5 ch.8, 2-byte elements).
// clang-23 toolchain: 6-arg builtin (g0 x4, g1 x8, g2 x4, g3 x4, extra x8, cpol)
DEV void tdm_load_2d(unsigned lds_off, const void* gptr,
                     unsigned tensor_d0, unsigned tensor_d1,
                     unsigned long long stride0,
                     unsigned tile_d0, unsigned tile_d1) {
  unsigned long long ga = (unsigned long long)gptr;
  uint32x4 g0;
  g0[0] = 1u;                                        // count=1, user mode
  g0[1] = lds_off;                                   // lds_addr
  g0[2] = (unsigned)(ga & 0xffffffffull);            // global_addr[31:0]
  g0[3] = (unsigned)((ga >> 32) & 0x1ffffffull)      // global_addr[56:32]
          | (2u << 30);                              // type = 2 ("image")
  int32x8 g1;
  g1[0] = (int)(1u << 16);                           // data_size=1 (2 bytes)
  g1[1] = (int)((tensor_d0 & 0xffffu) << 16);        // tensor_dim0[15:0]
  g1[2] = (int)(((tensor_d0 >> 16) & 0xffffu) |
                ((tensor_d1 & 0xffffu) << 16));      // dim0[31:16] | dim1[15:0]
  g1[3] = (int)(((tensor_d1 >> 16) & 0xffffu) |
                ((tile_d0 & 0xffffu) << 16));        // dim1[31:16] | tile_dim0
  g1[4] = (int)(tile_d1 & 0xffffu);                  // tile_dim1 (tile_dim2=0)
  g1[5] = (int)(unsigned)(stride0 & 0xffffffffull);  // dim0_stride[31:0]
  g1[6] = (int)(unsigned)((stride0 >> 32) & 0xffffull); // stride[47:32]
  g1[7] = 0;
  int32x4 z4; z4[0] = 0; z4[1] = 0; z4[2] = 0; z4[3] = 0;   // groups 2/3 (2D: unused)
  int32x8 z8;
#pragma unroll
  for (int i = 0; i < 8; ++i) z8[i] = 0;
  __builtin_amdgcn_tensor_load_to_lds(g0, g1, z4, z4, z8, 0);
}
#else
#define USE_TDM 0
#endif

// ---------------------------------------------------------------- weights -> bf16
__global__ void cvt_bf16_kernel(const float* __restrict__ src,
                                unsigned short* __restrict__ dst, int n) {
  int i = blockIdx.x * blockDim.x + threadIdx.x;
  for (; i < n; i += gridDim.x * blockDim.x) dst[i] = f2bf(src[i]);
}

// ---------------------------------------------------------------- layernorm -> bf16
__global__ __launch_bounds__(256) void ln_kernel(const float* __restrict__ x,
                                                 const float* __restrict__ g,
                                                 const float* __restrict__ bta,
                                                 unsigned short* __restrict__ out) {
  __shared__ float red[256];
  int row = blockIdx.x;
  int tid = threadIdx.x;
  const float* xr = x + (size_t)row * Dc;
  float4 vv = *(const float4*)(xr + tid * 4);
  float va[4] = {vv.x, vv.y, vv.z, vv.w};

  float s = va[0] + va[1] + va[2] + va[3];
  red[tid] = s; __syncthreads();
  for (int o = 128; o > 0; o >>= 1) { if (tid < o) red[tid] += red[tid + o]; __syncthreads(); }
  float mean = red[0] / (float)Dc;
  __syncthreads();

  float ss = 0.f;
  for (int e = 0; e < 4; ++e) { float d = va[e] - mean; ss += d * d; }
  red[tid] = ss; __syncthreads();
  for (int o = 128; o > 0; o >>= 1) { if (tid < o) red[tid] += red[tid + o]; __syncthreads(); }
  float var = red[0] / (float)(Dc - 1);
  float inv = 1.0f / (sqrtf(var) + EPSc);

  int c = tid * 4;
  unsigned short* orow = out + (size_t)row * Dc + c;
  for (int e = 0; e < 4; ++e)
    orow[e] = f2bf(g[c + e] * (va[e] - mean) * inv + bta[c + e]);
}

// ---------------------------------------------------------------- bf16 WMMA GEMM
//  C[M,N] = act( A[M,K](bf16) x W[K,N](bf16) + bias [+ resid] )
template <int OUT_BF16, int RELU, int RESID>
__global__ __launch_bounds__(256) void gemm_bf16(const unsigned short* __restrict__ A,
                                                 const unsigned short* __restrict__ W,
                                                 const float* __restrict__ bias,
                                                 const float* __restrict__ resid,
                                                 float* __restrict__ outF,
                                                 unsigned short* __restrict__ outB,
                                                 int M, int N, int K) {
  constexpr int BM = 128, BN = 128, BK = 32;
  __shared__ unsigned short sA[BM][BK];   // [m][k] row-major (TDM destination)
  __shared__ unsigned short sB[BN][BK];   // W tile transposed: [n][k]

  int tid  = threadIdx.x;
  int lane = tid & 31;
  int wave = tid >> 5;             // 0..7
  int wy   = wave >> 2;            // 0..1  -> 64-row band
  int wx   = wave & 3;             // 0..3  -> 32-col band
  int half = lane >> 4;
  int l16  = lane & 15;
  int bm = blockIdx.y * BM;
  int bn = blockIdx.x * BN;

  CFrag acc[4][2];
#pragma unroll
  for (int mt = 0; mt < 4; ++mt)
#pragma unroll
    for (int nt = 0; nt < 2; ++nt)
#pragma unroll
      for (int e = 0; e < 8; ++e) acc[mt][nt].f[e] = 0.f;

  for (int k0 = 0; k0 < K; k0 += BK) {
#if USE_TDM
    // A tile (128x32 bf16, row-major) via Tensor Data Mover, issued by wave 0
    if (wave == 0)
      tdm_load_2d((unsigned)(unsigned long long)&sA[0][0],
                  A + (size_t)bm * K + k0,
                  (unsigned)BK, (unsigned)BM,
                  (unsigned long long)K, (unsigned)BK, (unsigned)BM);
#else
#pragma unroll
    for (int i = 0; i < 4; ++i) {
      int c = tid * 4 + i;                 // 0..1023
      int r = c >> 3;
      int kk = (c & 7) * 4;
      uint2 val = *(const uint2*)(A + (size_t)(bm + r) * K + k0 + kk);
      *(uint2*)&sA[r][kk] = val;
    }
#endif
    // stage W tile transposed into [n][k] (TDM cannot transpose)
#pragma unroll
    for (int i = 0; i < 4; ++i) {
      int c = tid * 4 + i;
      int kk = c >> 5;                      // 0..31
      int nn = (c & 31) * 4;
      uint2 val = *(const uint2*)(W + (size_t)(k0 + kk) * N + bn + nn);
      sB[nn + 0][kk] = (unsigned short)(val.x & 0xffffu);
      sB[nn + 1][kk] = (unsigned short)(val.x >> 16);
      sB[nn + 2][kk] = (unsigned short)(val.y & 0xffffu);
      sB[nn + 3][kk] = (unsigned short)(val.y >> 16);
    }
    if (k0 + BK < K)
      __builtin_prefetch(W + (size_t)(k0 + BK + (tid >> 3)) * N + bn, 0, 0);
#if USE_TDM
    if (wave == 0) __builtin_amdgcn_s_wait_tensorcnt(0);
#endif
    __syncthreads();

    AFrag bfr[2];
#pragma unroll
    for (int nt = 0; nt < 2; ++nt) {
      int n_local = wx * 32 + nt * 16 + l16;
#pragma unroll
      for (int j = 0; j < 8; ++j)
        bfr[nt].u[j] = *(const unsigned int*)&sB[n_local][half * 16 + 2 * j];
    }
#pragma unroll
    for (int mt = 0; mt < 4; ++mt) {
      AFrag af;
      int m_local = wy * 64 + mt * 16 + l16;
#pragma unroll
      for (int j = 0; j < 8; ++j)
        af.u[j] = *(const unsigned int*)&sA[m_local][a_kmap(j, half)];
#pragma unroll
      for (int nt = 0; nt < 2; ++nt)
        acc[mt][nt].v = __builtin_amdgcn_wmma_f32_16x16x32_bf16(
            false, af.v, false, bfr[nt].v, (short)0, acc[mt][nt].v, false, false);
    }
    __syncthreads();
  }

  // epilogue (C layout: row = r + 8*half, col = l16)
#pragma unroll
  for (int mt = 0; mt < 4; ++mt)
#pragma unroll
    for (int nt = 0; nt < 2; ++nt) {
      int n = bn + wx * 32 + nt * 16 + l16;
#pragma unroll
      for (int r = 0; r < 8; ++r) {
        int m = bm + wy * 64 + mt * 16 + r + 8 * half;
        float v = acc[mt][nt].f[r] + bias[n];
        if (RESID) v += resid[(size_t)m * N + n];
        if (RELU)  v = fmaxf(v, 0.f);
        if (OUT_BF16) outB[(size_t)m * N + n] = f2bf(v);
        else          outF[(size_t)m * N + n] = v;
      }
    }
}

// ---------------------------------------------------------------- flash attention
// q,k,v : [B,S,H,DK] bf16 ; mask [B,1,S] i32 ; attn out [B,S,D] bf16
__global__ __launch_bounds__(256) void attn_kernel(const unsigned short* __restrict__ q,
                                                   const unsigned short* __restrict__ k,
                                                   const unsigned short* __restrict__ v,
                                                   const int* __restrict__ mask,
                                                   unsigned short* __restrict__ attn) {
  __shared__ unsigned short sK[32][64];       // [kv][d] row-major (TDM destination)
  __shared__ unsigned short sV[64][32];       // [d][kv] (transposed)
  __shared__ unsigned short sP[8][16][32];    // per-wave softmax tile

  const float scale = 0.125f;                 // 1/sqrt(64)
  int b = blockIdx.z, h = blockIdx.y;
  int tid = threadIdx.x, lane = tid & 31, wave = tid >> 5;
  int half = lane >> 4, l16 = lane & 15;
  int qbase = blockIdx.x * 128 + wave * 16;

  // Q fragments: 16x64 = two 16x32 A-fragments
  AFrag qf[2];
#pragma unroll
  for (int f = 0; f < 2; ++f)
#pragma unroll
    for (int j = 0; j < 8; ++j) {
      int kd = f * 32 + a_kmap(j, half);
      size_t addr = ((size_t)((b * Sc + qbase + l16) * Hc + h)) * DKc + kd;
      qf[f].u[j] = *(const unsigned int*)(q + addr);
    }

  float mrow[8], lrow[8];
  CFrag oacc[4];
#pragma unroll
  for (int r = 0; r < 8; ++r) { mrow[r] = -1e30f; lrow[r] = 0.f; }
#pragma unroll
  for (int dt = 0; dt < 4; ++dt)
#pragma unroll
    for (int e = 0; e < 8; ++e) oacc[dt].f[e] = 0.f;

  for (int kc = 0; kc < Sc; kc += 32) {
#if USE_TDM
    // K tile (32 rows x 64 elems, row stride H*DK) via TDM
    if (wave == 0)
      tdm_load_2d((unsigned)(unsigned long long)&sK[0][0],
                  k + ((size_t)((b * Sc + kc) * Hc + h)) * DKc,
                  (unsigned)DKc, 32u,
                  (unsigned long long)(Hc * DKc), (unsigned)DKc, 32u);
#endif
    // cooperative stage of V (transposed; TDM cannot transpose) and K fallback
#pragma unroll
    for (int i = 0; i < 2; ++i) {
      int c = tid * 2 + i;                   // 0..511
      int srow = c >> 4;                     // 0..31
      int dd = (c & 15) * 4;
      size_t gk = ((size_t)((b * Sc + kc + srow) * Hc + h)) * DKc + dd;
#if !USE_TDM
      uint2 kv_ = *(const uint2*)(k + gk);
      *(uint2*)&sK[srow][dd] = kv_;
#endif
      uint2 vv_ = *(const uint2*)(v + gk);
      sV[dd + 0][srow] = (unsigned short)(vv_.x & 0xffffu);
      sV[dd + 1][srow] = (unsigned short)(vv_.x >> 16);
      sV[dd + 2][srow] = (unsigned short)(vv_.y & 0xffffu);
      sV[dd + 3][srow] = (unsigned short)(vv_.y >> 16);
    }
#if USE_TDM
    if (wave == 0) __builtin_amdgcn_s_wait_tensorcnt(0);
#endif
    __syncthreads();

    // scores: two 16x16 tiles over 32 kv columns
    CFrag sc[2];
#pragma unroll
    for (int nt = 0; nt < 2; ++nt) {
#pragma unroll
      for (int e = 0; e < 8; ++e) sc[nt].f[e] = 0.f;
#pragma unroll
      for (int f = 0; f < 2; ++f) {
        AFrag kb;
#pragma unroll
        for (int j = 0; j < 8; ++j)
          kb.u[j] = *(const unsigned int*)&sK[nt * 16 + l16][f * 32 + half * 16 + 2 * j];
        sc[nt].v = __builtin_amdgcn_wmma_f32_16x16x32_bf16(
            false, qf[f].v, false, kb.v, (short)0, sc[nt].v, false, false);
      }
    }

    int mk0 = mask[b * Sc + kc + l16];
    int mk1 = mask[b * Sc + kc + 16 + l16];
    float p0[8], p1[8], corr[8];
#pragma unroll
    for (int r = 0; r < 8; ++r) {
      p0[r] = (mk0 == 0) ? NEGc : sc[0].f[r] * scale;
      p1[r] = (mk1 == 0) ? NEGc : sc[1].f[r] * scale;
      float vmx = fmaxf(p0[r], p1[r]);
#pragma unroll
      for (int o = 1; o < 16; o <<= 1) vmx = fmaxf(vmx, __shfl_xor(vmx, o, 32));
      float mnew = fmaxf(mrow[r], vmx);
      corr[r] = __expf(mrow[r] - mnew);
      p0[r] = __expf(p0[r] - mnew);
      p1[r] = __expf(p1[r] - mnew);
      float rs = p0[r] + p1[r];
#pragma unroll
      for (int o = 1; o < 16; o <<= 1) rs += __shfl_xor(rs, o, 32);
      lrow[r] = lrow[r] * corr[r] + rs;
      mrow[r] = mnew;
    }
#pragma unroll
    for (int dt = 0; dt < 4; ++dt)
#pragma unroll
      for (int r = 0; r < 8; ++r) oacc[dt].f[r] *= corr[r];

    // reshape P (C layout -> A layout) through per-wave LDS
#pragma unroll
    for (int r = 0; r < 8; ++r) {
      sP[wave][r + 8 * half][l16]      = f2bf(p0[r]);
      sP[wave][r + 8 * half][16 + l16] = f2bf(p1[r]);
    }
    AFrag pf;
#pragma unroll
    for (int j = 0; j < 8; ++j)
      pf.u[j] = *(const unsigned int*)&sP[wave][l16][a_kmap(j, half)];

    // P(16x32) x V(32x64)
#pragma unroll
    for (int dt = 0; dt < 4; ++dt) {
      AFrag vb;
#pragma unroll
      for (int j = 0; j < 8; ++j)
        vb.u[j] = *(const unsigned int*)&sV[dt * 16 + l16][half * 16 + 2 * j];
      oacc[dt].v = __builtin_amdgcn_wmma_f32_16x16x32_bf16(
          false, pf.v, false, vb.v, (short)0, oacc[dt].v, false, false);
    }
    __syncthreads();
  }

  // epilogue: O / l, merge heads into [B,S,D]
#pragma unroll
  for (int dt = 0; dt < 4; ++dt) {
    int d = h * DKc + dt * 16 + l16;
#pragma unroll
    for (int r = 0; r < 8; ++r) {
      int row = qbase + r + 8 * half;
      float val = oacc[dt].f[r] / lrow[r];
      attn[((size_t)(b * Sc + row)) * Dc + d] = f2bf(val);
    }
  }
}

// ---------------------------------------------------------------- launcher
extern "C" void kernel_launch(void* const* d_in, const int* in_sizes, int n_in,
                              void* d_out, int out_size, void* d_ws, size_t ws_size,
                              hipStream_t stream) {
  (void)in_sizes; (void)n_in; (void)out_size; (void)ws_size;
  const float* x     = (const float*)d_in[0];
  const int*   mask  = (const int*)d_in[1];
  const float* wq    = (const float*)d_in[2];
  const float* bq    = (const float*)d_in[3];
  const float* wk    = (const float*)d_in[4];
  const float* bk    = (const float*)d_in[5];
  const float* wv    = (const float*)d_in[6];
  const float* bv    = (const float*)d_in[7];
  const float* wo    = (const float*)d_in[8];
  const float* bo    = (const float*)d_in[9];
  const float* ln1g  = (const float*)d_in[10];
  const float* ln1b  = (const float*)d_in[11];
  const float* ln2g  = (const float*)d_in[12];
  const float* ln2b  = (const float*)d_in[13];
  const float* w1    = (const float*)d_in[14];
  const float* b1    = (const float*)d_in[15];
  const float* w2    = (const float*)d_in[16];
  const float* b2    = (const float*)d_in[17];

  char* ws = (char*)d_ws;
  size_t off = 0;
  auto take = [&](size_t bytes) {
    char* p = ws + off;
    off += (bytes + 255) & ~(size_t)255;
    return p;
  };
  const size_t DD  = (size_t)Dc * Dc;          // 1M
  const size_t DDF = (size_t)Dc * DFFc;        // 4M
  const size_t ACT = (size_t)BS * Dc;          // 4M elems
  unsigned short* wqb = (unsigned short*)take(DD * 2);
  unsigned short* wkb = (unsigned short*)take(DD * 2);
  unsigned short* wvb = (unsigned short*)take(DD * 2);
  unsigned short* wob = (unsigned short*)take(DD * 2);
  unsigned short* w1b = (unsigned short*)take(DDF * 2);
  unsigned short* w2b = (unsigned short*)take(DDF * 2);
  unsigned short* xn  = (unsigned short*)take(ACT * 2);   // also reused as xn2
  unsigned short* qb  = (unsigned short*)take(ACT * 2);   // q,k,v,attn contiguous
  unsigned short* kb  = (unsigned short*)take(ACT * 2);
  unsigned short* vb  = (unsigned short*)take(ACT * 2);
  unsigned short* ab  = (unsigned short*)take(ACT * 2);
  float*          x1  = (float*)take(ACT * 4);
  unsigned short* hb  = qb;   // 32MB FFN hidden aliases q/k/v/attn (all consumed)
  unsigned short* xn2 = xn;   // aliases xn (consumed by QKV GEMMs)

  dim3 blk(256);
  // 1. weights -> bf16
  cvt_bf16_kernel<<<4096, blk, 0, stream>>>(wq, wqb, (int)DD);
  cvt_bf16_kernel<<<4096, blk, 0, stream>>>(wk, wkb, (int)DD);
  cvt_bf16_kernel<<<4096, blk, 0, stream>>>(wv, wvb, (int)DD);
  cvt_bf16_kernel<<<4096, blk, 0, stream>>>(wo, wob, (int)DD);
  cvt_bf16_kernel<<<8192, blk, 0, stream>>>(w1, w1b, (int)DDF);
  cvt_bf16_kernel<<<8192, blk, 0, stream>>>(w2, w2b, (int)DDF);

  // 2. LN1
  ln_kernel<<<BS, blk, 0, stream>>>(x, ln1g, ln1b, xn);

  // 3. QKV projections
  dim3 gP(Dc / 128, BS / 128);                  // (8,32)
  gemm_bf16<1, 0, 0><<<gP, blk, 0, stream>>>(xn, wqb, bq, nullptr, nullptr, qb, BS, Dc, Dc);
  gemm_bf16<1, 0, 0><<<gP, blk, 0, stream>>>(xn, wkb, bk, nullptr, nullptr, kb, BS, Dc, Dc);
  gemm_bf16<1, 0, 0><<<gP, blk, 0, stream>>>(xn, wvb, bv, nullptr, nullptr, vb, BS, Dc, Dc);

  // 4. attention
  dim3 gA(Sc / 128, Hc, Bc);                    // (16,16,2)
  attn_kernel<<<gA, blk, 0, stream>>>(qb, kb, vb, mask, ab);

  // 5. output projection + residual -> x1 (fp32)
  gemm_bf16<0, 0, 1><<<gP, blk, 0, stream>>>(ab, wob, bo, x, x1, nullptr, BS, Dc, Dc);

  // 6. LN2
  ln_kernel<<<BS, blk, 0, stream>>>(x1, ln2g, ln2b, xn2);

  // 7. FFN1 + ReLU -> h (bf16)
  dim3 gF1(DFFc / 128, BS / 128);               // (32,32)
  gemm_bf16<1, 1, 0><<<gF1, blk, 0, stream>>>(xn2, w1b, b1, nullptr, nullptr, hb, BS, DFFc, Dc);

  // 8. FFN2 + residual -> d_out (fp32)
  gemm_bf16<0, 0, 1><<<gP, blk, 0, stream>>>(hb, w2b, b2, x1, (float*)d_out, nullptr, BS, Dc, DFFc);
}
